// LMU_43808666419810
// MI455X (gfx1250) — matrix operand order repaired and checked by reading, not compiled, not run on hardware
//
#include <hip/hip_runtime.h>
#include <cstdint>

// ---------------------------------------------------------------------------
// LMU layer for MI455X (gfx1250): chunked linear-recurrence scan + bf16 WMMA
// GEMMs. Wave32, 16x16x32 bf16 WMMA, f32 accumulate.
// ---------------------------------------------------------------------------

typedef __attribute__((ext_vector_type(16))) __bf16 v16bf;
typedef __attribute__((ext_vector_type(8)))  float  v8f;

union FragAB { v16bf v; uint4 u[2]; };

constexpr int BATCH = 64;
constexpr int SEQ   = 4096;
constexpr int IND   = 128;
constexpr int MD    = 32;
constexpr int ORD   = 64;
constexpr int HID   = 512;
constexpr int KDIM  = MD * ORD;   // 2048
constexpr int L     = 64;         // chunk length
constexpr int CHK   = SEQ / L;    // 64 chunks

__device__ __forceinline__ unsigned short f2bf(float f) {
  unsigned int u = __float_as_uint(f);
  u += 0x7FFFu + ((u >> 16) & 1u);          // round-to-nearest-even
  return (unsigned short)(u >> 16);
}

// ---------------------------------------------------------------------------
// Kernel P: Pvec[k] = B @ A^k (k=0..L-1) and AL = A^L.  Single workgroup.
// ---------------------------------------------------------------------------
__global__ __launch_bounds__(256) void lmu_prep_powers(
    const float* __restrict__ A, const float* __restrict__ Bv,
    float* __restrict__ Pvec, float* __restrict__ AL) {
  __shared__ float sA[ORD * ORD];
  __shared__ float cur[ORD * ORD];
  __shared__ float nxt[ORD * ORD];
  __shared__ float sB[ORD];
  int tid = threadIdx.x;
  for (int i = tid; i < ORD * ORD; i += 256) {
    sA[i] = A[i];
    cur[i] = ((i >> 6) == (i & 63)) ? 1.0f : 0.0f;   // identity
  }
  if (tid < ORD) sB[tid] = Bv[tid];
  __syncthreads();
  for (int k = 0; k < L; ++k) {
    if (tid < ORD) {                       // Pvec[k] = B @ cur (cur = A^k)
      float s = 0.0f;
      for (int p = 0; p < ORD; ++p) s += sB[p] * cur[p * ORD + tid];
      Pvec[k * ORD + tid] = s;
    }
    for (int i = tid; i < ORD * ORD; i += 256) {   // nxt = cur @ A
      int r = i >> 6, c = i & 63;
      float s = 0.0f;
      for (int p = 0; p < ORD; ++p) s += cur[r * ORD + p] * sA[p * ORD + c];
      nxt[i] = s;
    }
    __syncthreads();
    for (int i = tid; i < ORD * ORD; i += 256) cur[i] = nxt[i];
    __syncthreads();
  }
  for (int i = tid; i < ORD * ORD; i += 256) AL[i] = cur[i];
}

// ---------------------------------------------------------------------------
// Kernel E: u = x @ kernel  (memory-bound; x read once, coalesced float4)
// ---------------------------------------------------------------------------
__global__ __launch_bounds__(256) void lmu_encode(
    const float* __restrict__ x, const float* __restrict__ kern,
    float* __restrict__ u) {
  __shared__ float sk[IND * MD];   // 16 KB
  __shared__ float sx[8 * IND];    // 4 KB
  int tid = threadIdx.x, g = tid >> 5, lane = tid & 31;
  for (int i = tid; i < IND * MD; i += 256) sk[i] = kern[i];
  long row = (long)blockIdx.x * 8 + g;
  float4 xv = ((const float4*)(x + row * IND))[lane];
  ((float4*)(sx + g * IND))[lane] = xv;
  __syncthreads();
  float s = 0.0f;
  for (int i = 0; i < IND; ++i) s += sx[g * IND + i] * sk[i * MD + lane];
  u[row * MD + lane] = s;          // md = lane (0..31)
}

// ---------------------------------------------------------------------------
// Kernel W: dst[n*K+k] = bf16(src[k*N+n])  (transpose + convert, one-time)
// ---------------------------------------------------------------------------
__global__ void lmu_transpose_bf16(const float* __restrict__ src,
                                   unsigned short* __restrict__ dst,
                                   int K, int N) {
  long idx = (long)blockIdx.x * blockDim.x + threadIdx.x;
  if (idx >= (long)K * N) return;
  int k = (int)(idx % K), n = (int)(idx / K);
  dst[(long)n * K + k] = f2bf(src[(long)k * N + n]);
}

// ---------------------------------------------------------------------------
// Kernel S: S[b,c] = sum_{j=1..L} u[b,cL+j-1,:] outer Pvec[L-j]
// ---------------------------------------------------------------------------
__global__ __launch_bounds__(256) void lmu_chunk_sums(
    const float* __restrict__ u, const float* __restrict__ Pvec,
    float* __restrict__ S) {
  __shared__ float sP[L * ORD];   // 16 KB
  __shared__ float su[L * MD];    // 8 KB
  int c = blockIdx.x, b = blockIdx.y, tid = threadIdx.x;
  for (int i = tid; i < L * ORD; i += 256) sP[i] = Pvec[i];
  const float* ub = u + ((long)b * SEQ + (long)c * L) * MD;
  for (int i = tid; i < L * MD; i += 256) su[i] = ub[i];
  __syncthreads();
  int o = tid & 63, mg = tid >> 6;
  for (int md = mg; md < MD; md += 4) {
    float s = 0.0f;
    for (int j = 1; j <= L; ++j)
      s += su[(j - 1) * MD + md] * sP[(L - j) * ORD + o];
    S[(((long)b * CHK + c) * MD + md) * ORD + o] = s;
  }
}

// ---------------------------------------------------------------------------
// Kernel M: boundary scan  M_{c+1} = M_c @ A^L + S_c  (64 serial steps, per b)
// ---------------------------------------------------------------------------
__global__ __launch_bounds__(256) void lmu_boundary_scan(
    const float* __restrict__ S, const float* __restrict__ AL,
    float* __restrict__ Mb) {
  __shared__ float sAL[ORD * ORD];
  __shared__ float cur[MD * ORD];
  __shared__ float nxt[MD * ORD];
  __shared__ float sS[MD * ORD];
  int b = blockIdx.x, tid = threadIdx.x;
  for (int i = tid; i < ORD * ORD; i += 256) sAL[i] = AL[i];
  for (int i = tid; i < MD * ORD; i += 256) cur[i] = 0.0f;
  __syncthreads();
  for (int c = 0; c < CHK; ++c) {
    float* mb = Mb + ((long)b * CHK + c) * (MD * ORD);
    for (int i = tid; i < MD * ORD; i += 256) mb[i] = cur[i];
    const float* sc = S + ((long)b * CHK + c) * (MD * ORD);
    for (int i = tid; i < MD * ORD; i += 256) sS[i] = sc[i];
    __syncthreads();
    int o = tid & 63, mg = tid >> 6;
    for (int md = mg; md < MD; md += 4) {
      float s = sS[md * ORD + o];
      for (int p = 0; p < ORD; ++p) s += cur[md * ORD + p] * sAL[p * ORD + o];
      nxt[md * ORD + o] = s;
    }
    __syncthreads();
    for (int i = tid; i < MD * ORD; i += 256) cur[i] = nxt[i];
    __syncthreads();
  }
}

// ---------------------------------------------------------------------------
// Kernel Z: per (b, chunk) replay the 64-step recurrence from the boundary
// state into a 256 KB LDS bf16 hin tile (dynamic LDS; 320 KB WGP), then
// z = hin @ Wh via v_wmma_f32_16x16x32_bf16, written straight to d_out.
// ---------------------------------------------------------------------------
__global__ __launch_bounds__(256) void lmu_chunk_gemm(
    const float* __restrict__ A, const float* __restrict__ Bv,
    const float* __restrict__ u, const float* __restrict__ Mb,
    const unsigned short* __restrict__ WhT, float* __restrict__ out) {
  extern __shared__ char smem[];
  float* sA  = (float*)smem;            // 4096 f
  float* sBv = sA + ORD * ORD;          // 64 f
  float* su  = sBv + ORD;               // 2048 f
  float* m0  = su + L * MD;             // 2048 f
  float* m1  = m0 + MD * ORD;           // 2048 f
  unsigned short* hin = (unsigned short*)(m1 + MD * ORD);  // 64 x 2048 bf16

  int c = blockIdx.x, b = blockIdx.y, tid = threadIdx.x;
  for (int i = tid; i < ORD * ORD; i += 256) sA[i] = A[i];
  if (tid < ORD) sBv[tid] = Bv[tid];
  const float* ub = u + ((long)b * SEQ + (long)c * L) * MD;
  for (int i = tid; i < L * MD; i += 256) su[i] = ub[i];
  const float* mb = Mb + ((long)b * CHK + c) * (MD * ORD);
  for (int i = tid; i < MD * ORD; i += 256) m0[i] = mb[i];
  __syncthreads();

  // Phase 1: m_j = m_{j-1} @ A + u_j * B ; hin row j-1 = bf16(vec(m_j))
  float* cur = m0; float* nxt = m1;
  int o = tid & 63, mg = tid >> 6;
  for (int j = 1; j <= L; ++j) {
    for (int md = mg; md < MD; md += 4) {
      float s = su[(j - 1) * MD + md] * sBv[o];
      for (int p = 0; p < ORD; ++p) s += cur[md * ORD + p] * sA[p * ORD + o];
      nxt[md * ORD + o] = s;
      hin[(j - 1) * KDIM + md * ORD + o] = f2bf(s);
    }
    __syncthreads();
    float* t = cur; cur = nxt; nxt = t;
  }

  // Phase 2: (64 x 2048) @ (2048 x 512) bf16 WMMA, 4 M-tiles x 32 N-tiles
  int wave = tid >> 5, lane = tid & 31;
  int arow = lane & 15;
  int rhalf = (lane < 16) ? 0 : 8;             // D M-offset
  int boff  = (lane < 16) ? 0 : 16;            // B K-offset
  float* orow = out + ((long)b * SEQ + (long)c * L) * HID;
  for (int tile = wave; tile < (L / 16) * (HID / 16); tile += 8) {
    int mtile = tile & 3, ntile = tile >> 2;
    const unsigned short* wb = WhT + (long)(ntile * 16 + arow) * KDIM;
    const unsigned short* ar = hin + (mtile * 16 + arow) * KDIM;
    v8f acc = {};
    for (int ks = 0; ks < KDIM / 32; ++ks) {
      int kb = ks * 32;
      __builtin_prefetch(wb + kb + 256, 0, 0);   // global_prefetch_b8
      FragAB a, bf;
      a.u[0]  = *(const uint4*)(ar + kb + boff);
      a.u[1]  = *(const uint4*)(ar + kb + boff + 8);
      bf.u[0] = *(const uint4*)(wb + kb + boff);
      bf.u[1] = *(const uint4*)(wb + kb + boff + 8);
      acc = __builtin_amdgcn_wmma_f32_16x16x32_bf16(
          false, a.v, false, bf.v, (short)0, acc, false, false);
    }
    for (int i = 0; i < 8; ++i) {
      int m = i + rhalf;
      orow[(long)(mtile * 16 + m) * HID + ntile * 16 + arow] = acc[i];
    }
  }
}

// ---------------------------------------------------------------------------
// Kernel H: serial hidden scan, 16 batch rows per WG (4 WGs, independent).
// Per step: acc = h_prev @ Uh (M=16 WMMA), h = tanh(acc + z + bh); z read
// from d_out and overwritten in place with h.
// ---------------------------------------------------------------------------
__global__ __launch_bounds__(512) void lmu_hidden_scan(
    const unsigned short* __restrict__ UhT, const float* __restrict__ bh,
    float* __restrict__ out) {
  __shared__ unsigned short hprev[16 * HID];  // 16 KB bf16
  __shared__ float sbh[HID];
  int tid = threadIdx.x;
  long bbase = (long)blockIdx.x * 16;
  for (int i = tid; i < 16 * HID; i += 512) hprev[i] = 0;
  for (int i = tid; i < HID; i += 512) sbh[i] = bh[i];
  __syncthreads();

  int wave = tid >> 5, lane = tid & 31;
  int nt0 = wave * 2, nt1 = wave * 2 + 1;     // 16 waves x 2 = 32 N-tiles
  int arow = lane & 15;
  int rhalf = (lane < 16) ? 0 : 8;
  int boff  = (lane < 16) ? 0 : 16;
  const unsigned short* wb0 = UhT + (long)(nt0 * 16 + arow) * HID;
  const unsigned short* wb1 = UhT + (long)(nt1 * 16 + arow) * HID;

  for (int t = 0; t < SEQ; ++t) {
    v8f acc0 = {}, acc1 = {};
    for (int ks = 0; ks < HID / 32; ++ks) {
      int kb = ks * 32;
      FragAB a, b0, b1;
      const unsigned short* ap = hprev + arow * HID + kb + boff;
      a.u[0]  = *(const uint4*)ap;
      a.u[1]  = *(const uint4*)(ap + 8);
      b0.u[0] = *(const uint4*)(wb0 + kb + boff);
      b0.u[1] = *(const uint4*)(wb0 + kb + boff + 8);
      b1.u[0] = *(const uint4*)(wb1 + kb + boff);
      b1.u[1] = *(const uint4*)(wb1 + kb + boff + 8);
      acc0 = __builtin_amdgcn_wmma_f32_16x16x32_bf16(
          false, a.v, false, b0.v, (short)0, acc0, false, false);
      acc1 = __builtin_amdgcn_wmma_f32_16x16x32_bf16(
          false, a.v, false, b1.v, (short)0, acc1, false, false);
    }
    __syncthreads();   // all hprev reads for step t complete
    for (int i = 0; i < 8; ++i) {
      int r = i + rhalf;
      int c0 = nt0 * 16 + arow;
      float* z0 = out + ((bbase + r) * SEQ + t) * HID + c0;
      float v0 = tanhf(acc0[i] + *z0 + sbh[c0]);
      *z0 = v0;
      hprev[r * HID + c0] = f2bf(v0);
      int c1 = nt1 * 16 + arow;
      float* z1 = out + ((bbase + r) * SEQ + t) * HID + c1;
      float v1 = tanhf(acc1[i] + *z1 + sbh[c1]);
      *z1 = v1;
      hprev[r * HID + c1] = f2bf(v1);
    }
    __syncthreads();
  }
}

// ---------------------------------------------------------------------------
extern "C" void kernel_launch(void* const* d_in, const int* in_sizes, int n_in,
                              void* d_out, int out_size, void* d_ws, size_t ws_size,
                              hipStream_t stream) {
  const float* x    = (const float*)d_in[0];
  const float* kern = (const float*)d_in[1];
  const float* A    = (const float*)d_in[2];
  const float* Bv   = (const float*)d_in[3];
  const float* Wh   = (const float*)d_in[4];
  const float* Uh   = (const float*)d_in[5];
  const float* bh   = (const float*)d_in[6];
  float* out = (float*)d_out;

  char* ws = (char*)d_ws;
  size_t off = 0;
  auto carve = [&](size_t bytes) -> char* {
    char* p = ws + off;
    off = (off + bytes + 255) & ~(size_t)255;
    return p;
  };
  float* u    = (float*)carve((size_t)BATCH * SEQ * MD * 4);        // 32 MB
  float* Pvec = (float*)carve((size_t)L * ORD * 4);                 // 16 KB
  float* AL   = (float*)carve((size_t)ORD * ORD * 4);               // 16 KB
  float* S    = (float*)carve((size_t)BATCH * CHK * MD * ORD * 4);  // 32 MB
  float* Mb   = (float*)carve((size_t)BATCH * CHK * MD * ORD * 4);  // 32 MB
  unsigned short* WhT = (unsigned short*)carve((size_t)KDIM * HID * 2); // 2 MB
  unsigned short* UhT = (unsigned short*)carve((size_t)HID * HID * 2);  // 0.5 MB

  lmu_prep_powers<<<1, 256, 0, stream>>>(A, Bv, Pvec, AL);
  lmu_encode<<<(BATCH * SEQ) / 8, 256, 0, stream>>>(x, kern, u);
  {
    long tot = (long)KDIM * HID;
    lmu_transpose_bf16<<<(int)((tot + 255) / 256), 256, 0, stream>>>(Wh, WhT, KDIM, HID);
  }
  {
    long tot = (long)HID * HID;
    lmu_transpose_bf16<<<(int)((tot + 255) / 256), 256, 0, stream>>>(Uh, UhT, HID, HID);
  }
  lmu_chunk_sums<<<dim3(CHK, BATCH), 256, 0, stream>>>(u, Pvec, S);
  lmu_boundary_scan<<<BATCH, 256, 0, stream>>>(S, AL, Mb);
  size_t smemZ = (size_t)(ORD * ORD + ORD + L * MD + 2 * MD * ORD) * 4
               + (size_t)L * KDIM * 2;   // ~296 KB of the 320 KB WGP LDS
  lmu_chunk_gemm<<<dim3(CHK, BATCH), 256, smemZ, stream>>>(A, Bv, u, Mb, WhT, out);
  lmu_hidden_scan<<<BATCH / 16, 512, 0, stream>>>(UhT, bh, out);
}
